// SwarmGNN_43971875176945
// MI455X (gfx1250) — compile-verified, matrix operand order checked by reading, not run on hardware
//
#include <hip/hip_runtime.h>
#include <hip/hip_bf16.h>

#define NN    50000
#define EE    640000
#define ET    (EE + NN)      // 690000 edges incl. self loops
#define BB    64
#define HID   128
#define NH    4
#define CH    32
#define GOALD 64
#define LRS   0.2f

typedef _Float16 v8h  __attribute__((ext_vector_type(8)));
typedef _Float16 v16h __attribute__((ext_vector_type(16)));
typedef float    v8f  __attribute__((ext_vector_type(8)));

static inline int cdiv_h(long long a, long long b) { return (int)((a + b - 1) / b); }

// native fp32 global atomic add (GLOBAL_ATOMIC_ADD_F32), no CAS loop
__device__ __forceinline__ void atomAddF32(float* p, float v) {
  unsafeAtomicAdd(p, v);
}

// ---------------------------------------------------------------------------
// monotone float<->uint encoding for atomic segment-max over signed floats
// ---------------------------------------------------------------------------
__device__ __forceinline__ unsigned fenc(float f) {
  unsigned u = __float_as_uint(f);
  return (u & 0x80000000u) ? ~u : (u | 0x80000000u);
}
__device__ __forceinline__ float fdec(unsigned e) {
  unsigned u = (e & 0x80000000u) ? (e ^ 0x80000000u) : ~e;
  return __uint_as_float(u);
}

__device__ __forceinline__ void edge_sd(int e, const int* __restrict__ ei, int& s, int& d) {
  if (e < EE) { s = ei[e]; d = ei[EE + e]; }
  else        { s = e - EE; d = e - EE; }
}

__device__ __forceinline__ float wave_sum32(float v) {
  #pragma unroll
  for (int m = 16; m >= 1; m >>= 1) v += __shfl_xor(v, m, 32);
  return v;
}

// ---------------------------------------------------------------------------
// WMMA GEMM: Y[M x 128] = X[M x K](f16) * W + bias,  Wt[128 x K] = W^T (f16)
// block = 256 threads = 8 wave32; wave w computes columns [16w, 16w+16)
// K must be a multiple of 32. One 16x16 D tile per wave, K swept by x32 WMMA.
// ---------------------------------------------------------------------------
__global__ __launch_bounds__(256) void gemm_f16_wmma(
    const _Float16* __restrict__ X, const _Float16* __restrict__ Wt,
    const float* __restrict__ bias, float* __restrict__ Y, int M, int K)
{
  const int wave   = threadIdx.x >> 5;
  const int lane   = threadIdx.x & 31;
  const int laneHi = lane >> 4;
  const int l15    = lane & 15;
  const int m0     = blockIdx.x * 16;
  const int nb     = wave * 16;

  int arow = m0 + l15; if (arow >= M) arow = M - 1;
  const _Float16* __restrict__ xrow = X  + (size_t)arow * K;
  const _Float16* __restrict__ wrow = Wt + (size_t)(nb + l15) * K;

  v8f acc = {};
  for (int k0 = 0; k0 < K; k0 += 32) {
    v8h aLo = *(const v8h*)(xrow + k0 +      8 * laneHi);
    v8h aHi = *(const v8h*)(xrow + k0 + 16 + 8 * laneHi);
    v8h bLo = *(const v8h*)(wrow + k0 +      8 * laneHi);
    v8h bHi = *(const v8h*)(wrow + k0 + 16 + 8 * laneHi);
    v16h a = __builtin_shufflevector(aLo, aHi, 0,1,2,3,4,5,6,7,8,9,10,11,12,13,14,15);
    v16h b = __builtin_shufflevector(bLo, bHi, 0,1,2,3,4,5,6,7,8,9,10,11,12,13,14,15);
    acc = __builtin_amdgcn_wmma_f32_16x16x32_f16(false, a, false, b, (short)0, acc,
                                                 false, false);
  }
  const int col = nb + l15;
  const float bv = bias[col];
  #pragma unroll
  for (int r = 0; r < 8; ++r) {
    int m = m0 + r + 8 * laneHi;
    if (m < M) Y[(size_t)m * HID + col] = acc[r] + bv;
  }
}

// ---------------------------------------------------------------------------
// weight convert + transpose + K-pad:  Wt[n*Kp + k] = (k<K0) ? W[k*128 + n] : 0
// ---------------------------------------------------------------------------
__global__ void wt_conv(const float* __restrict__ W, _Float16* __restrict__ Wt,
                        int K0, int Kp)
{
  int idx = blockIdx.x * blockDim.x + threadIdx.x;
  if (idx >= 128 * Kp) return;
  int n = idx / Kp, k = idx % Kp;
  Wt[(size_t)n * Kp + k] = (k < K0) ? (_Float16)W[(size_t)k * 128 + n] : (_Float16)0.0f;
}

// activation f32 -> f16 with zero K padding
__global__ void act_conv(const float* __restrict__ src, _Float16* __restrict__ dst,
                         int M, int Ks, int Kp)
{
  int idx = blockIdx.x * blockDim.x + threadIdx.x;
  if (idx >= M * Kp) return;
  int m = idx / Kp, k = idx % Kp;
  dst[idx] = (k < Ks) ? (_Float16)src[(size_t)m * Ks + k] : (_Float16)0.0f;
}

// ---------------------------------------------------------------------------
// wave-per-row LayerNorm(+ReLU) over 128 features; also emits f16 copy
// ---------------------------------------------------------------------------
__global__ __launch_bounds__(256) void ln_relu(
    float* __restrict__ Y, const float* __restrict__ g, const float* __restrict__ b,
    _Float16* __restrict__ Y16, int M)
{
  int row = blockIdx.x * 8 + (threadIdx.x >> 5);
  if (row >= M) return;
  int l = threadIdx.x & 31;
  float v[4], s = 0.f;
  #pragma unroll
  for (int j = 0; j < 4; ++j) { v[j] = Y[(size_t)row * HID + j * 32 + l]; s += v[j]; }
  float mu = wave_sum32(s) * (1.0f / HID);
  float vs = 0.f;
  #pragma unroll
  for (int j = 0; j < 4; ++j) { float d = v[j] - mu; vs += d * d; }
  float inv = rsqrtf(wave_sum32(vs) * (1.0f / HID) + 1e-5f);
  #pragma unroll
  for (int j = 0; j < 4; ++j) {
    int c = j * 32 + l;
    float y = fmaxf((v[j] - mu) * inv * g[c] + b[c], 0.f);
    Y[(size_t)row * HID + c] = y;
    Y16[(size_t)row * HID + c] = (_Float16)y;
  }
}

// node-type select + type embedding
__global__ void select_type(const float* __restrict__ hc, const float* __restrict__ hg,
                            const float* __restrict__ temb, const int* __restrict__ nt,
                            float* __restrict__ h, _Float16* __restrict__ h16)
{
  int idx = blockIdx.x * blockDim.x + threadIdx.x;
  if (idx >= NN * HID) return;
  int n = idx / HID, c = idx % HID, t = nt[n];
  float v = ((t == 0) ? hc[idx] : (t == 1) ? hg[idx] : 0.f) + temb[t * HID + c];
  h[idx] = v; h16[idx] = (_Float16)v;
}

// ---------------------------------------------------------------------------
// GATv2 edge pipeline
// ---------------------------------------------------------------------------
__global__ void edge_logits(const float* __restrict__ xl, const float* __restrict__ xr,
                            const float* __restrict__ att, const int* __restrict__ ei,
                            float* __restrict__ logits, unsigned* __restrict__ mEnc)
{
  int idx = blockIdx.x * blockDim.x + threadIdx.x;
  if (idx >= ET * NH) return;
  int e = idx >> 2, hh = idx & 3;
  int s, d; edge_sd(e, ei, s, d);
  const float* pl = xl + (size_t)s * HID + hh * CH;
  const float* pr = xr + (size_t)d * HID + hh * CH;
  const float* pa = att + hh * CH;
  float acc = 0.f;
  #pragma unroll 8
  for (int c = 0; c < CH; ++c) {
    float x = pl[c] + pr[c];
    x = (x > 0.f) ? x : LRS * x;
    acc += x * pa[c];
  }
  logits[idx] = acc;
  atomicMax(&mEnc[d * NH + hh], fenc(acc));
}

__global__ void edge_exp(const float* __restrict__ logits, const unsigned* __restrict__ mEnc,
                         const int* __restrict__ ei, float* __restrict__ exb,
                         float* __restrict__ den)
{
  int idx = blockIdx.x * blockDim.x + threadIdx.x;
  if (idx >= ET * NH) return;
  int e = idx >> 2, hh = idx & 3;
  int s, d; edge_sd(e, ei, s, d);
  float ex = __expf(logits[idx] - fdec(mEnc[d * NH + hh]));
  exb[idx] = ex;
  atomAddF32(&den[d * NH + hh], ex);
}

__global__ void edge_accum(const float* __restrict__ exb, const float* __restrict__ xl,
                           const int* __restrict__ ei, float* __restrict__ agg)
{
  long long idx = (long long)blockIdx.x * blockDim.x + threadIdx.x;
  if (idx >= (long long)ET * 32) return;
  int e = (int)(idx >> 5), l = (int)(idx & 31);
  int s, d; edge_sd(e, ei, s, d);
  #pragma unroll
  for (int hh = 0; hh < NH; ++hh) {
    float w = exb[e * NH + hh];
    atomAddF32(&agg[(size_t)d * HID + hh * CH + l], w * xl[(size_t)s * HID + hh * CH + l]);
  }
}

// residual + LN + FiLM + ReLU  (wave per node), emits f16 copy for next layer
__global__ __launch_bounds__(256) void node_update(
    float* __restrict__ h, _Float16* __restrict__ h16,
    const float* __restrict__ agg, const float* __restrict__ den,
    const float* __restrict__ gbias, const float* __restrict__ lng,
    const float* __restrict__ lnb, const float* __restrict__ gammaB,
    const float* __restrict__ betaB, const int* __restrict__ batch)
{
  int n = blockIdx.x * 8 + (threadIdx.x >> 5);
  if (n >= NN) return;
  int l = threadIdx.x & 31;
  int bi = batch[n];
  float v[4], s = 0.f;
  #pragma unroll
  for (int j = 0; j < 4; ++j) {
    int c = j * 32 + l;
    v[j] = h[(size_t)n * HID + c] + agg[(size_t)n * HID + c] / den[n * NH + j] + gbias[c];
    s += v[j];
  }
  float mu = wave_sum32(s) * (1.0f / HID);
  float vs = 0.f;
  #pragma unroll
  for (int j = 0; j < 4; ++j) { float d = v[j] - mu; vs += d * d; }
  float inv = rsqrtf(wave_sum32(vs) * (1.0f / HID) + 1e-5f);
  #pragma unroll
  for (int j = 0; j < 4; ++j) {
    int c = j * 32 + l;
    float y = (v[j] - mu) * inv * lng[c] + lnb[c];
    float ga = gammaB[bi * HID + c] + 1.0f;
    float be = betaB[bi * HID + c];
    y = fmaxf(y * ga + be, 0.f);
    h[(size_t)n * HID + c] = y;
    h16[(size_t)n * HID + c] = (_Float16)y;
  }
}

// ---------------------------------------------------------------------------
// graph pooling: mean via f32 atomic add, max via unsigned-bit atomic (h >= 0)
// ---------------------------------------------------------------------------
__global__ void pool_scatter(const float* __restrict__ h, const int* __restrict__ batch,
                             float* __restrict__ meanacc, unsigned* __restrict__ maxacc,
                             int* __restrict__ cnt)
{
  long long idx = (long long)blockIdx.x * blockDim.x + threadIdx.x;
  if (idx >= (long long)NN * 32) return;
  int n = (int)(idx >> 5), l = (int)(idx & 31);
  int b = batch[n];
  #pragma unroll
  for (int j = 0; j < 4; ++j) {
    int c = j * 32 + l;
    float v = h[(size_t)n * HID + c];
    atomAddF32(&meanacc[b * HID + c], v);
    atomicMax(&maxacc[b * HID + c], __float_as_uint(v));  // valid: v >= 0 post-ReLU
  }
  if (l == 0) atomicAdd(&cnt[b], 1);
}

__global__ void pool_finalize(const float* __restrict__ meanacc,
                              const unsigned* __restrict__ maxacc,
                              const int* __restrict__ cnt, _Float16* __restrict__ gcat16)
{
  int idx = blockIdx.x * blockDim.x + threadIdx.x;
  if (idx >= BB * 2 * HID) return;
  int b = idx / (2 * HID), c = idx % (2 * HID);
  float v = (c < HID) ? meanacc[b * HID + c] / fmaxf((float)cnt[b], 1.f)
                      : __uint_as_float(maxacc[b * HID + (c - HID)]);
  gcat16[idx] = (_Float16)v;
}

__global__ void relu_to_f16(const float* __restrict__ Y, _Float16* __restrict__ Y16, int n)
{
  int idx = blockIdx.x * blockDim.x + threadIdx.x;
  if (idx < n) Y16[idx] = (_Float16)fmaxf(Y[idx], 0.f);
}

// ---------------------------------------------------------------------------
extern "C" void kernel_launch(void* const* d_in, const int* in_sizes, int n_in,
                              void* d_out, int out_size, void* d_ws, size_t ws_size,
                              hipStream_t stream) {
  // ---- inputs ----
  const float* x      = (const float*)d_in[0];            // [N,25]
  const float* goal   = (const float*)d_in[1];            // [B,64]
  const int*   ei     = (const int*)d_in[2];              // [2,E]
  const int*   ntype  = (const int*)d_in[3];              // [N]
  const int*   batch  = (const int*)d_in[4];              // [N]
  const float* cw1 = (const float*)d_in[5];  const float* cb1 = (const float*)d_in[6];
  const float* cg1 = (const float*)d_in[7];  const float* cbt1= (const float*)d_in[8];
  const float* cw2 = (const float*)d_in[9];  const float* cb2 = (const float*)d_in[10];
  const float* cg2 = (const float*)d_in[11]; const float* cbt2= (const float*)d_in[12];
  const float* cw3 = (const float*)d_in[13]; const float* cb3 = (const float*)d_in[14];
  const float* gw1 = (const float*)d_in[15]; const float* gb1 = (const float*)d_in[16];
  const float* gg1 = (const float*)d_in[17]; const float* gbt1= (const float*)d_in[18];
  const float* gw2 = (const float*)d_in[19]; const float* gb2 = (const float*)d_in[20];
  const float* temb= (const float*)d_in[21];
  const float* wl  = (const float*)d_in[22]; const float* bl  = (const float*)d_in[23];
  const float* wr  = (const float*)d_in[24]; const float* br  = (const float*)d_in[25];
  const float* att = (const float*)d_in[26]; const float* gbias=(const float*)d_in[27];
  const float* lng = (const float*)d_in[28]; const float* lnb = (const float*)d_in[29];
  const float* fgw = (const float*)d_in[30]; const float* fgb = (const float*)d_in[31];
  const float* fbw = (const float*)d_in[32]; const float* fbb = (const float*)d_in[33];
  const float* ow1 = (const float*)d_in[34]; const float* ob1 = (const float*)d_in[35];
  const float* ow2 = (const float*)d_in[36]; const float* ob2 = (const float*)d_in[37];

  // ---- workspace layout ----
  char* base = (char*)d_ws; size_t off = 0;
  auto alloc = [&](size_t bytes) -> char* {
    char* p = base + off; off += (bytes + 255) & ~(size_t)255; return p;
  };
  float*     h      = (float*)alloc((size_t)NN * HID * 4);
  _Float16*  h16    = (_Float16*)alloc((size_t)NN * HID * 2);
  float*     xlb    = (float*)alloc((size_t)NN * HID * 4);
  float*     xrb    = (float*)alloc((size_t)NN * HID * 4);
  float*     agg    = (float*)alloc((size_t)NN * HID * 4);
  _Float16*  tmp16  = (_Float16*)alloc((size_t)NN * HID * 2);
  _Float16*  Xp     = (_Float16*)alloc((size_t)NN * 32 * 2);
  float*     logits = (float*)alloc((size_t)ET * NH * 4);
  float*     exb    = (float*)alloc((size_t)ET * NH * 4);
  unsigned*  mEnc   = (unsigned*)alloc((size_t)NN * NH * 4);
  float*     den    = (float*)alloc((size_t)NN * NH * 4);
  _Float16*  cw1t = (_Float16*)alloc(128 * 32 * 2);
  _Float16*  cw2t = (_Float16*)alloc(128 * 128 * 2);
  _Float16*  cw3t = (_Float16*)alloc(128 * 128 * 2);
  _Float16*  gw1t = (_Float16*)alloc(128 * 32 * 2);
  _Float16*  gw2t = (_Float16*)alloc(128 * 128 * 2);
  _Float16*  wlt  = (_Float16*)alloc(3 * 128 * 128 * 2);
  _Float16*  wrt  = (_Float16*)alloc(3 * 128 * 128 * 2);
  _Float16*  fgwt = (_Float16*)alloc(3 * 128 * 64 * 2);
  _Float16*  fbwt = (_Float16*)alloc(3 * 128 * 64 * 2);
  _Float16*  ow1t = (_Float16*)alloc(128 * 256 * 2);
  _Float16*  ow2t = (_Float16*)alloc(128 * 128 * 2);
  _Float16*  goal16 = (_Float16*)alloc(BB * GOALD * 2);
  float*     gammaB = (float*)alloc(BB * HID * 4);
  float*     betaB  = (float*)alloc(BB * HID * 4);
  int*       cnt    = (int*)alloc(BB * 4);
  float*     meanacc= (float*)alloc(BB * HID * 4);
  unsigned*  maxacc = (unsigned*)alloc(BB * HID * 4);
  _Float16*  gcat16 = (_Float16*)alloc(BB * 2 * HID * 2);
  float*     pt     = (float*)alloc(BB * HID * 4);
  _Float16*  pt16   = (_Float16*)alloc(BB * HID * 2);

  auto WCONV = [&](const float* W, _Float16* Wt, int K0, int Kp) {
    wt_conv<<<cdiv_h(128 * Kp, 256), 256, 0, stream>>>(W, Wt, K0, Kp);
  };
  auto GEMM = [&](const _Float16* X, const _Float16* Wt, const float* bias,
                  float* Y, int M, int K) {
    gemm_f16_wmma<<<cdiv_h(M, 16), 256, 0, stream>>>(X, Wt, bias, Y, M, K);
  };

  // ---- weight conversion (f32 -> transposed, K-padded f16) ----
  WCONV(cw1, cw1t, 25, 32);   WCONV(cw2, cw2t, 128, 128); WCONV(cw3, cw3t, 128, 128);
  WCONV(gw1, gw1t, 12, 32);   WCONV(gw2, gw2t, 128, 128);
  for (int i = 0; i < 3; ++i) {
    WCONV(wl  + (size_t)i * 128 * 128, wlt  + (size_t)i * 128 * 128, 128, 128);
    WCONV(wr  + (size_t)i * 128 * 128, wrt  + (size_t)i * 128 * 128, 128, 128);
    WCONV(fgw + (size_t)i * 64  * 128, fgwt + (size_t)i * 128 * 64,  64,  64);
    WCONV(fbw + (size_t)i * 64  * 128, fbwt + (size_t)i * 128 * 64,  64,  64);
  }
  WCONV(ow1, ow1t, 256, 256); WCONV(ow2, ow2t, 128, 128);
  act_conv<<<cdiv_h((size_t)NN * 32, 256), 256, 0, stream>>>(x, Xp, NN, 25, 32);
  act_conv<<<cdiv_h(BB * GOALD, 256), 256, 0, stream>>>(goal, goal16, BB, GOALD, GOALD);

  // ---- node encoders (agg/xl/xr borrowed as scratch) ----
  GEMM(Xp, cw1t, cb1, xlb, NN, 32);
  ln_relu<<<cdiv_h(NN, 8), 256, 0, stream>>>(xlb, cg1, cbt1, tmp16, NN);
  GEMM(tmp16, cw2t, cb2, xrb, NN, 128);
  ln_relu<<<cdiv_h(NN, 8), 256, 0, stream>>>(xrb, cg2, cbt2, tmp16, NN);
  GEMM(tmp16, cw3t, cb3, agg, NN, 128);                 // hc -> agg
  GEMM(Xp, gw1t, gb1, xlb, NN, 32);
  ln_relu<<<cdiv_h(NN, 8), 256, 0, stream>>>(xlb, gg1, gbt1, tmp16, NN);
  GEMM(tmp16, gw2t, gb2, xrb, NN, 128);                 // hg -> xr
  select_type<<<cdiv_h((size_t)NN * HID, 256), 256, 0, stream>>>(agg, xrb, temb, ntype, h, h16);

  // ---- GATv2 + FiLM layers ----
  for (int i = 0; i < 3; ++i) {
    GEMM(h16, wlt + (size_t)i * 128 * 128, bl + i * HID, xlb, NN, 128);
    GEMM(h16, wrt + (size_t)i * 128 * 128, br + i * HID, xrb, NN, 128);
    hipMemsetAsync(mEnc, 0, (size_t)NN * NH * 4, stream);
    hipMemsetAsync(den,  0, (size_t)NN * NH * 4, stream);
    hipMemsetAsync(agg,  0, (size_t)NN * HID * 4, stream);
    const float* att_l = att + i * NH * CH;
    edge_logits<<<cdiv_h((size_t)ET * NH, 256), 256, 0, stream>>>(xlb, xrb, att_l, ei, logits, mEnc);
    edge_exp<<<cdiv_h((size_t)ET * NH, 256), 256, 0, stream>>>(logits, mEnc, ei, exb, den);
    edge_accum<<<cdiv_h((size_t)ET * 32, 256), 256, 0, stream>>>(exb, xlb, ei, agg);
    GEMM(goal16, fgwt + (size_t)i * 128 * 64, fgb + i * HID, gammaB, BB, 64);
    GEMM(goal16, fbwt + (size_t)i * 128 * 64, fbb + i * HID, betaB,  BB, 64);
    node_update<<<cdiv_h(NN, 8), 256, 0, stream>>>(h, h16, agg, den, gbias + i * HID,
                                                   lng + i * HID, lnb + i * HID,
                                                   gammaB, betaB, batch);
  }

  // ---- pooling + output projection ----
  hipMemsetAsync(meanacc, 0, BB * HID * 4, stream);
  hipMemsetAsync(maxacc,  0, BB * HID * 4, stream);
  hipMemsetAsync(cnt,     0, BB * 4, stream);
  pool_scatter<<<cdiv_h((size_t)NN * 32, 256), 256, 0, stream>>>(h, batch, meanacc, maxacc, cnt);
  pool_finalize<<<cdiv_h(BB * 2 * HID, 256), 256, 0, stream>>>(meanacc, maxacc, cnt, gcat16);
  GEMM(gcat16, ow1t, ob1, pt, BB, 256);
  relu_to_f16<<<cdiv_h(BB * HID, 256), 256, 0, stream>>>(pt, pt16, BB * HID);
  GEMM(pt16, ow2t, ob2, (float*)d_out, BB, 128);        // g -> d_out[0 : B*128)

  // second output: node features h
  hipMemcpyAsync((float*)d_out + (size_t)BB * HID, h, (size_t)NN * HID * 4,
                 hipMemcpyDeviceToDevice, stream);
}
